// SparseStructureAttention_3685081940020
// MI455X (gfx1250) — matrix operand order repaired and checked by reading, not compiled, not run on hardware
//
#include <hip/hip_runtime.h>
#include <hip/hip_bf16.h>
#include <stdint.h>

// Problem constants (from reference)
#define NRES   8192
#define KNB    32
#define DMODEL 768
#define CPAIR  128
#define NHEAD  8
#define SDIM   64
#define PPTS   8
#define NPTS   24                 // 3*P
#define QKVC   (NHEAD * 3 * SDIM) // 1536
#define PTSC   (NHEAD * NPTS * 3) // 576
#define FEATC  1792               // H*CP + H*S + H*P*3 + H*P

#define LDSTR  40                 // padded LDS stride (ushorts): 80B -> bank-conflict free

typedef unsigned short ushort_t;
typedef __attribute__((ext_vector_type(16))) __bf16         v16bf;
typedef __attribute__((ext_vector_type(8)))  float          v8f;
typedef __attribute__((ext_vector_type(4)))  int            v4i;

#define AS_GLOBAL __attribute__((address_space(1)))
#define AS_LDS    __attribute__((address_space(3)))

#if __has_builtin(__builtin_amdgcn_global_load_async_to_lds_b128)
#define USE_ASYNC_LDS 1
#endif

__device__ __forceinline__ unsigned short f2bf(float f) {
  unsigned int u = __builtin_bit_cast(unsigned int, f);
  u += 0x7fffu + ((u >> 16) & 1u);   // round-to-nearest-even
  return (unsigned short)(u >> 16);
}

// Fragment-order permutation of a 32-wide K chunk (ISA 7.12.2 16-bit A layout):
// positions [0..7]=K0..7, [8..15]=K16..23, [16..23]=K8..15, [24..31]=K24..31.
__device__ __forceinline__ int perm32(int c) {
  return (c & ~31) | ((c & 8) << 1) | ((c & 16) >> 1) | (c & 7);
}

__device__ __forceinline__ float wsum(float v) {
#pragma unroll
  for (int o = 16; o > 0; o >>= 1) v += __shfl_xor(v, o, 32);
  return v;
}
__device__ __forceinline__ float wmaxr(float v) {
#pragma unroll
  for (int o = 16; o > 0; o >>= 1) v = fmaxf(v, __shfl_xor(v, o, 32));
  return v;
}

// -------------------------------------------- pack W (f32 KxN) -> B tiles (bf16)
// Bp[((k/32)*Ncol + n)*32 + perm32(k&31)] = bf16(W[k*Ncol + n])  (K-major per column, fragment order)
__global__ __launch_bounds__(256)
void pack_b_kernel(const float* __restrict__ W, ushort_t* __restrict__ Bp,
                   int Ncol, int Kdim) {
  int idx = blockIdx.x * blockDim.x + threadIdx.x;
  int total = (Kdim >> 5) * Ncol;
  if (idx >= total) return;
  int kt = idx / Ncol;
  int n  = idx - kt * Ncol;
  union { ushort_t s[32]; uint4 q[4]; } buf;
#pragma unroll
  for (int o = 0; o < 32; ++o)
    buf.s[o] = f2bf(W[(size_t)(kt * 32 + perm32(o)) * Ncol + n]);
  uint4* dst = (uint4*)&Bp[((size_t)kt * Ncol + n) * 32];
#pragma unroll
  for (int j = 0; j < 4; ++j) dst[j] = buf.q[j];
}

// ------------------------------------------------- LN(local) + frames_from_pos
// x written bf16 in fragment (perm32) column order for direct WMMA-A staging.
__global__ __launch_bounds__(256)
void prep_kernel(const float* __restrict__ local, const float* __restrict__ pos,
                 const float* __restrict__ lns, const float* __restrict__ lnb,
                 ushort_t* __restrict__ xbf, float* __restrict__ Rm, float* __restrict__ tv) {
  const int n = blockIdx.x;
  const int tid = threadIdx.x;
  const int lane = tid & 31, wave = tid >> 5;
  __shared__ float sred[8], qred[8];
  __shared__ float s_mean, s_rstd;

  const float* row = local + (size_t)n * DMODEL;
  float v0 = row[tid], v1 = row[tid + 256], v2 = row[tid + 512];
  float s = v0 + v1 + v2;
  float q = v0 * v0 + v1 * v1 + v2 * v2;
  s = wsum(s); q = wsum(q);
  if (lane == 0) { sred[wave] = s; qred[wave] = q; }
  __syncthreads();
  if (tid == 0) {
    float S = 0.f, Q = 0.f;
    for (int i = 0; i < 8; ++i) { S += sred[i]; Q += qred[i]; }
    float m   = S * (1.f / DMODEL);
    float var = Q * (1.f / DMODEL) - m * m;
    s_mean = m; s_rstd = rsqrtf(var + 1e-5f);
  }
  __syncthreads();
  float m = s_mean, r = s_rstd;
  ushort_t* xr = xbf + (size_t)n * DMODEL;
  xr[perm32(tid)]       = f2bf((v0 - m) * r * lns[tid]       + lnb[tid]);
  xr[perm32(tid + 256)] = f2bf((v1 - m) * r * lns[tid + 256] + lnb[tid + 256]);
  xr[perm32(tid + 512)] = f2bf((v2 - m) * r * lns[tid + 512] + lnb[tid + 512]);

  if (tid == 0) {  // frames: R = [e1 e2 e3] stored as 3 row-vectors, t = CA
    const float* p = pos + (size_t)n * 42;
    float Nx=p[0],Ny=p[1],Nz=p[2], Ax=p[3],Ay=p[4],Az=p[5], Cx=p[6],Cy=p[7],Cz=p[8];
    float v1x=Cx-Ax, v1y=Cy-Ay, v1z=Cz-Az;
    float v2x=Nx-Ax, v2y=Ny-Ay, v2z=Nz-Az;
    float i1 = rsqrtf(v1x*v1x + v1y*v1y + v1z*v1z + 1e-8f);
    float e1x=v1x*i1, e1y=v1y*i1, e1z=v1z*i1;
    float d = v2x*e1x + v2y*e1y + v2z*e1z;
    float u2x=v2x-d*e1x, u2y=v2y-d*e1y, u2z=v2z-d*e1z;
    float i2 = rsqrtf(u2x*u2x + u2y*u2y + u2z*u2z + 1e-8f);
    float e2x=u2x*i2, e2y=u2y*i2, e2z=u2z*i2;
    float e3x=e1y*e2z-e1z*e2y, e3y=e1z*e2x-e1x*e2z, e3z=e1x*e2y-e1y*e2x;
    float* Rr = Rm + (size_t)n * 9;
    Rr[0]=e1x; Rr[1]=e1y; Rr[2]=e1z;
    Rr[3]=e2x; Rr[4]=e2y; Rr[5]=e2z;
    Rr[6]=e3x; Rr[7]=e3y; Rr[8]=e3z;
    float* tr = tv + (size_t)n * 3;
    tr[0]=Ax; tr[1]=Ay; tr[2]=Az;
  }
}

// -------------------------------------------------- bf16 WMMA GEMM  C = A*B(+b)
// A: MxK bf16, rows pre-permuted (perm32) -> fragment-contiguous 32-K chunks.
// Bp: pre-packed K-major tiles.  Block 256 thr = 8 waves -> 128x64 C tile;
// wave -> 32x32 patch (4 WMMAs).  Double-buffered LDS; on CDNA5 the staging is
// GLOBAL_LOAD_ASYNC_TO_LDS_B128 (ASYNCcnt) pipelined one K-tile ahead.
__global__ __launch_bounds__(256)
void gemm_bf16_wmma(const ushort_t* __restrict__ A, const ushort_t* __restrict__ Bp,
                    float* __restrict__ C, const float* __restrict__ biasv,
                    int Ncol, int Kdim) {
  __shared__ __align__(16) ushort_t As[2][128 * LDSTR];
  __shared__ __align__(16) ushort_t Bs[2][64 * LDSTR];
  const int tid  = threadIdx.x;
  const int lane = tid & 31;
  const int wave = tid >> 5;
  const int m0 = blockIdx.y * 128;
  const int n0 = blockIdx.x * 64;
  const int mq = wave & 3;    // 32-row slice of block
  const int nh = wave >> 2;   // 32-col slice of block
  const int nk = Kdim >> 5;

  v8f acc[2][2] = {};

  // per-thread staging chunks (b128 = 8 ushorts each)
  // A: chunks q = 2*tid+{0,1}: row = q>>2 (0..127), seg = q&3
  // B: chunk  tid: col = tid>>2 (0..63), seg = tid&3
  const int ar0 = (2 * tid) >> 2, as0 = (2 * tid) & 3;
  const int ar1 = (2 * tid + 1) >> 2, as1 = (2 * tid + 1) & 3;
  const int bcol = tid >> 2, bseg = tid & 3;

  auto issue_tile = [&](int kt, int buf) {
    const ushort_t* a0 = &A[(size_t)(m0 + ar0) * Kdim + kt * 32 + as0 * 8];
    const ushort_t* a1 = &A[(size_t)(m0 + ar1) * Kdim + kt * 32 + as1 * 8];
    const ushort_t* b0 = &Bp[((size_t)kt * Ncol + n0 + bcol) * 32 + bseg * 8];
    ushort_t* da0 = &As[buf][ar0 * LDSTR + as0 * 8];
    ushort_t* da1 = &As[buf][ar1 * LDSTR + as1 * 8];
    ushort_t* db0 = &Bs[buf][bcol * LDSTR + bseg * 8];
#ifdef USE_ASYNC_LDS
    __builtin_amdgcn_global_load_async_to_lds_b128(
        (AS_GLOBAL v4i*)a0, (AS_LDS v4i*)da0, 0, 0);
    __builtin_amdgcn_global_load_async_to_lds_b128(
        (AS_GLOBAL v4i*)a1, (AS_LDS v4i*)da1, 0, 0);
    __builtin_amdgcn_global_load_async_to_lds_b128(
        (AS_GLOBAL v4i*)b0, (AS_LDS v4i*)db0, 0, 0);
#else
    *(uint4*)da0 = *(const uint4*)a0;
    *(uint4*)da1 = *(const uint4*)a1;
    *(uint4*)db0 = *(const uint4*)b0;
#endif
  };

  issue_tile(0, 0);
  for (int kt = 0; kt < nk; ++kt) {
    const int buf = kt & 1;
#ifdef USE_ASYNC_LDS
    asm volatile("s_wait_asynccnt 0x0" ::: "memory");  // this wave's tile-kt copies done
#endif
    __syncthreads();                                    // whole block's tile kt visible
    if (kt + 1 < nk) issue_tile(kt + 1, buf ^ 1);       // stream next tile while computing

    union FragU { uint4 q[2]; v16bf b; };
    FragU fa[2], fb[2];
    const int half = (lane < 16) ? 0 : 16;
    const int l15  = lane & 15;
#pragma unroll
    for (int i = 0; i < 2; ++i) {
      const int row = mq * 32 + i * 16 + l15;
      fa[i].q[0] = *(const uint4*)&As[buf][row * LDSTR + half];
      fa[i].q[1] = *(const uint4*)&As[buf][row * LDSTR + half + 8];
      const int col = nh * 32 + i * 16 + l15;
      fb[i].q[0] = *(const uint4*)&Bs[buf][col * LDSTR + half];
      fb[i].q[1] = *(const uint4*)&Bs[buf][col * LDSTR + half + 8];
    }
#pragma unroll
    for (int i = 0; i < 2; ++i)
#pragma unroll
      for (int j = 0; j < 2; ++j)
        acc[i][j] = __builtin_amdgcn_wmma_f32_16x16x32_bf16(
            false, fa[i].b, false, fb[j].b, (short)0, acc[i][j], false, false);
    __syncthreads();   // protect buf from overwrite 2 iterations later
  }

  // C/D layout: lanes 0-15 rows 0..7 (vgpr r), lanes 16-31 rows 8..15.
  const int rbase = m0 + mq * 32 + ((lane < 16) ? 0 : 8);
  const int cbase = n0 + nh * 32 + (lane & 15);
#pragma unroll
  for (int j = 0; j < 2; ++j) {
    const int colj = cbase + j * 16;
    const float bb = biasv ? biasv[colj] : 0.f;
#pragma unroll
    for (int i = 0; i < 2; ++i)
#pragma unroll
      for (int r = 0; r < 8; ++r)
        C[(size_t)(rbase + i * 16 + r) * Ncol + colj] = acc[i][j][r] + bb;
  }
}

// ------------------------------------------------------- per-head LN on q and k
__global__ __launch_bounds__(256)
void qk_ln_kernel(float* __restrict__ qkv,
                  const float* __restrict__ qs, const float* __restrict__ qb,
                  const float* __restrict__ ks, const float* __restrict__ kb) {
  int idx = blockIdx.x * blockDim.x + threadIdx.x;
  if (idx >= NRES * NHEAD * 2) return;
  int n = idx >> 4;
  int r = idx & 15;
  int h = r >> 1;
  int isk = r & 1;
  float4* base = (float4*)(qkv + (size_t)n * QKVC + h * (3 * SDIM) + isk * SDIM);
  const float* sc = isk ? ks : qs;
  const float* bo = isk ? kb : qb;
  float4 v[16];
  float s = 0.f, q = 0.f;
#pragma unroll
  for (int c = 0; c < 16; ++c) {
    float4 x = base[c]; v[c] = x;
    s += x.x + x.y + x.z + x.w;
    q += x.x*x.x + x.y*x.y + x.z*x.z + x.w*x.w;
  }
  float m   = s * (1.f / SDIM);
  float var = q * (1.f / SDIM) - m * m;
  float rs  = rsqrtf(var + 1e-5f);
#pragma unroll
  for (int c = 0; c < 16; ++c) {
    float4 x = v[c];
    x.x = (x.x - m) * rs * sc[4*c+0] + bo[4*c+0];
    x.y = (x.y - m) * rs * sc[4*c+1] + bo[4*c+1];
    x.z = (x.z - m) * rs * sc[4*c+2] + bo[4*c+2];
    x.w = (x.w - m) * rs * sc[4*c+3] + bo[4*c+3];
    base[c] = x;
  }
}

// --------------------------------------------- rotate raw points to global frame
__global__ __launch_bounds__(256)
void rot_pts_kernel(const float* __restrict__ raw, const float* __restrict__ Rm,
                    const float* __restrict__ tv, float* __restrict__ ptsR) {
  int idx = blockIdx.x * blockDim.x + threadIdx.x;
  if (idx >= NRES * NHEAD * NPTS) return;
  int n = idx / (NHEAD * NPTS);
  int rem = idx - n * (NHEAD * NPTS);
  const float* src = raw + (size_t)n * PTSC + rem * 3;
  const float* R = Rm + (size_t)n * 9;
  const float* t = tv + (size_t)n * 3;
  float x = src[0], y = src[1], z = src[2];
  float* dst = ptsR + (size_t)n * PTSC + rem * 3;
  dst[0] = x * R[0] + y * R[3] + z * R[6] + t[0];
  dst[1] = x * R[1] + y * R[4] + z * R[7] + t[1];
  dst[2] = x * R[2] + y * R[5] + z * R[8] + t[2];
}

// ------------------------------------ fused gather-attention (1 block / residue)
// wave == head (8 waves), lane == neighbour k (wave32 == K exactly).
// feats written bf16 in perm32 column order (WMMA-A fragment layout).
__global__ __launch_bounds__(256)
void attn_kernel(const float* __restrict__ qkv, const float* __restrict__ ptsR,
                 const float* __restrict__ pair, const unsigned char* __restrict__ pmask,
                 const int* __restrict__ nbr, const float* __restrict__ Wb,
                 const float* __restrict__ gamma, const float* __restrict__ Rm,
                 const float* __restrict__ tv, ushort_t* __restrict__ feats) {
  __shared__ float pairS[KNB * CPAIR];     // 16 KB pair tile
  __shared__ float attnS[NHEAD][KNB];
  __shared__ float opS[NHEAD][NPTS];
  __shared__ int   nbS[KNB];

  const int n = blockIdx.x;
  const int tid = threadIdx.x;
  const int lane = tid & 31;
  const int h = tid >> 5;

  if (tid < KNB) {
    int nb = nbr[n * KNB + tid];
    nbS[tid] = nb < 0 ? 0 : nb;
  }
  for (int i = tid; i < KNB * CPAIR / 4; i += 256)
    ((float4*)pairS)[i] = ((const float4*)(pair + (size_t)n * KNB * CPAIR))[i];
  __syncthreads();

  int  nbraw = nbr[n * KNB + lane];
  bool valid = (pmask[n * KNB + lane] != 0) && (nbraw != -1);
  int  nb    = nbraw < 0 ? 0 : nbraw;

  // dot = sqrt(1/S) * q . k[nb]   (q uniform -> scalar loads, k via b128)
  const float4* q4 = (const float4*)(qkv + (size_t)n * QKVC + h * (3 * SDIM));
  const float4* k4 = (const float4*)(qkv + (size_t)nb * QKVC + h * (3 * SDIM) + SDIM);
  float dot = 0.f;
#pragma unroll
  for (int c = 0; c < 16; ++c) {
    float4 a = q4[c], b = k4[c];
    dot += a.x*b.x + a.y*b.y + a.z*b.z + a.w*b.w;
  }
  dot *= 0.125f;

  // dist = softplus(gamma)*wC/2 * sum |q_g - k_g|^2
  const float4* qg4 = (const float4*)(ptsR + (size_t)n * PTSC + h * (NPTS * 3));
  const float4* kg4 = (const float4*)(ptsR + (size_t)nb * PTSC + h * (NPTS * 3) + PPTS * 3);
  float dist = 0.f;
#pragma unroll
  for (int j = 0; j < 6; ++j) {
    float4 a = qg4[j], b = kg4[j];
    float dx = a.x-b.x, dy = a.y-b.y, dz = a.z-b.z, dw = a.w-b.w;
    dist += dx*dx + dy*dy + dz*dz + dw*dw;
  }
  float dfac = log1pf(__expf(gamma[h])) * (0.16666667f * 0.5f);  // wC = sqrt(2/72)
  dist *= dfac;

  // bias = pair[n,k,:] @ W_bias[:,h]   (pair from LDS, W_bias uniform)
  float bias = 0.f;
  {
    const float4* p4 = (const float4*)&pairS[lane * CPAIR];
#pragma unroll 8
    for (int c = 0; c < 32; ++c) {
      float4 pv = p4[c];
      bias += pv.x * Wb[(4*c+0) * NHEAD + h] + pv.y * Wb[(4*c+1) * NHEAD + h]
            + pv.z * Wb[(4*c+2) * NHEAD + h] + pv.w * Wb[(4*c+3) * NHEAD + h];
    }
  }

  const float wL = 0.57735027f;  // sqrt(1/3)
  float logit = valid ? wL * (dot + bias - dist) : -1e9f;
  float mx = wmaxr(logit);
  float e  = __expf(logit - mx);
  float ss = wsum(e);
  float at = valid ? e / ss : 0.f;
  attnS[h][lane] = at;
  __syncthreads();

  ushort_t* frow = feats + (size_t)n * FEATC;
  // out_pair (H,CP)
#pragma unroll
  for (int c4 = 0; c4 < 4; ++c4) {
    int c = lane + 32 * c4;
    float acc = 0.f;
    for (int k = 0; k < KNB; ++k) acc += attnS[h][k] * pairS[k * CPAIR + c];
    frow[perm32(h * CPAIR + c)] = f2bf(acc);
  }
  // out_scalar (H,S) from gathered v (L2-resident)
#pragma unroll
  for (int c2 = 0; c2 < 2; ++c2) {
    int c = lane + 32 * c2;
    float acc = 0.f;
    for (int k = 0; k < KNB; ++k)
      acc += attnS[h][k] * qkv[(size_t)nbS[k] * QKVC + h * (3 * SDIM) + 2 * SDIM + c];
    frow[perm32(NHEAD * CPAIR + h * SDIM + c)] = f2bf(acc);
  }
  // op (global frame) from gathered v_g
  if (lane < NPTS) {
    float acc = 0.f;
    for (int k = 0; k < KNB; ++k)
      acc += attnS[h][k] * ptsR[(size_t)nbS[k] * PTSC + h * (NPTS * 3) + 2 * PPTS * 3 + lane];
    opS[h][lane] = acc;
  }
  __syncthreads();

  const float* R = Rm + (size_t)n * 9;
  const float* t = tv + (size_t)n * 3;
  if (lane < NPTS) {  // rotate back: op_i = e_i . (op - t)
    int p = lane / 3, i = lane - 3 * p;
    float x = opS[h][p * 3 + 0] - t[0];
    float y = opS[h][p * 3 + 1] - t[1];
    float z = opS[h][p * 3 + 2] - t[2];
    float o = x * R[i * 3 + 0] + y * R[i * 3 + 1] + z * R[i * 3 + 2];
    frow[perm32(NHEAD * (CPAIR + SDIM) + h * (PPTS * 3) + lane)] = f2bf(o);
  }
  if (lane < PPTS) {  // norm is rotation-invariant
    float x = opS[h][lane * 3 + 0] - t[0];
    float y = opS[h][lane * 3 + 1] - t[1];
    float z = opS[h][lane * 3 + 2] - t[2];
    frow[perm32(NHEAD * (CPAIR + SDIM) + NHEAD * PPTS * 3 + h * PPTS + lane)] =
        f2bf(sqrtf(x * x + y * y + z * z + 1e-8f));
  }
}

// ------------------------------------------------------------------------------
extern "C" void kernel_launch(void* const* d_in, const int* in_sizes, int n_in,
                              void* d_out, int out_size, void* d_ws, size_t ws_size,
                              hipStream_t stream) {
  (void)in_sizes; (void)n_in; (void)out_size; (void)ws_size;
  const float* local       = (const float*)d_in[0];
  const float* pos         = (const float*)d_in[1];
  const float* pair        = (const float*)d_in[2];
  const unsigned char* pm  = (const unsigned char*)d_in[3];
  const int*   neighbours  = (const int*)d_in[4];
  const float* ln_local_s  = (const float*)d_in[9];
  const float* ln_local_b  = (const float*)d_in[10];
  const float* W_qkv       = (const float*)d_in[11];
  const float* ln_q_s      = (const float*)d_in[12];
  const float* ln_q_b      = (const float*)d_in[13];
  const float* ln_k_s      = (const float*)d_in[14];
  const float* ln_k_b      = (const float*)d_in[15];
  const float* W_pts       = (const float*)d_in[16];
  const float* W_bias      = (const float*)d_in[17];
  const float* gamma       = (const float*)d_in[18];
  const float* W_out       = (const float*)d_in[19];
  const float* b_out       = (const float*)d_in[20];
  float* out = (float*)d_out;

  char* ws = (char*)d_ws;
  size_t off = 0;
  auto alloc = [&](size_t b) { void* p = ws + off; off += (b + 255) & ~(size_t)255; return p; };

  ushort_t* xbf    = (ushort_t*)alloc((size_t)NRES * DMODEL * 2);
  ushort_t* WqkvB  = (ushort_t*)alloc((size_t)DMODEL * QKVC * 2);
  ushort_t* WptsB  = (ushort_t*)alloc((size_t)DMODEL * PTSC * 2);
  ushort_t* WoutB  = (ushort_t*)alloc((size_t)FEATC * DMODEL * 2);
  float*    Rm     = (float*)alloc((size_t)NRES * 9 * 4);
  float*    tv     = (float*)alloc((size_t)NRES * 3 * 4);
  float*    qkv    = (float*)alloc((size_t)NRES * QKVC * 4);
  float*    rawp   = (float*)alloc((size_t)NRES * PTSC * 4);
  float*    ptsR   = (float*)alloc((size_t)NRES * PTSC * 4);
  ushort_t* featsB = (ushort_t*)alloc((size_t)NRES * FEATC * 2);

  pack_b_kernel<<<((DMODEL / 32) * QKVC + 255) / 256, 256, 0, stream>>>(
      W_qkv, WqkvB, QKVC, DMODEL);
  pack_b_kernel<<<((DMODEL / 32) * PTSC + 255) / 256, 256, 0, stream>>>(
      W_pts, WptsB, PTSC, DMODEL);
  pack_b_kernel<<<((FEATC / 32) * DMODEL + 255) / 256, 256, 0, stream>>>(
      W_out, WoutB, DMODEL, FEATC);

  prep_kernel<<<NRES, 256, 0, stream>>>(local, pos, ln_local_s, ln_local_b, xbf, Rm, tv);

  gemm_bf16_wmma<<<dim3(QKVC / 64, NRES / 128), 256, 0, stream>>>(
      xbf, WqkvB, qkv, nullptr, QKVC, DMODEL);
  gemm_bf16_wmma<<<dim3(PTSC / 64, NRES / 128), 256, 0, stream>>>(
      xbf, WptsB, rawp, nullptr, PTSC, DMODEL);

  qk_ln_kernel<<<(NRES * NHEAD * 2 + 255) / 256, 256, 0, stream>>>(
      qkv, ln_q_s, ln_q_b, ln_k_s, ln_k_b);
  rot_pts_kernel<<<(NRES * NHEAD * NPTS + 255) / 256, 256, 0, stream>>>(rawp, Rm, tv, ptsR);

  attn_kernel<<<NRES, 256, 0, stream>>>(qkv, ptsR, pair, pm, neighbours, W_bias, gamma,
                                        Rm, tv, featsB);

  gemm_bf16_wmma<<<dim3(DMODEL / 64, NRES / 128), 256, 0, stream>>>(
      featsB, WoutB, out, b_out, DMODEL, FEATC);
}
